// Attention_64330020159669
// MI455X (gfx1250) — matrix operand order repaired and checked by reading, not compiled, not run on hardware
//
#include <hip/hip_runtime.h>
#include <hip/hip_bf16.h>

// ---------------------------------------------------------------------------
// MHA block for MI455X (gfx1250, wave32, WMMA).
// B=2, S=2048, E=2048, N=16 heads, H=128.
// All matmuls on v_wmma_f32_16x16x32_bf16; global->LDS staging uses the
// CDNA5 async-to-LDS path (ASYNCcnt) where the access pattern allows.
// ---------------------------------------------------------------------------

typedef __bf16 bf16;
typedef __attribute__((ext_vector_type(16))) bf16 v16bf;
typedef __attribute__((ext_vector_type(8)))  bf16 v8bf;
typedef __attribute__((ext_vector_type(4)))  bf16 v4bf;
typedef __attribute__((ext_vector_type(8)))  float v8f;

union V16 { v16bf v; v8bf h[2]; };

constexpr int Bz    = 2;
constexpr int SEQ   = 2048;
constexpr int EMB   = 2048;
constexpr int NHEAD = 16;
constexpr int HDIM  = 128;
constexpr int MROWS = Bz * SEQ;      // 4096
constexpr int NHH   = NHEAD * HDIM;  // 2048

__device__ __forceinline__ v8f wmma_bf16(v16bf a, v16bf b, v8f c) {
  // D = A(16x32 bf16) * B(32x16 bf16) + C(16x16 f32)
  return __builtin_amdgcn_wmma_f32_16x16x32_bf16(
      /*neg_a=*/false, a, /*neg_b=*/false, b,
      /*c_mod=*/(short)0, c, /*reuse_a=*/false, /*reuse_b=*/false);
}

// --- CDNA5 async global->LDS copies (tracked by ASYNCcnt) -------------------
__device__ __forceinline__ void async_ld_b64(unsigned lds_off, const void* g) {
  asm volatile("global_load_async_to_lds_b64 %0, %1, off"
               :: "v"(lds_off), "v"((unsigned long long)(uintptr_t)g)
               : "memory");
}
__device__ __forceinline__ void async_ld_b128(unsigned lds_off, const void* g) {
  asm volatile("global_load_async_to_lds_b128 %0, %1, off"
               :: "v"(lds_off), "v"((unsigned long long)(uintptr_t)g)
               : "memory");
}
__device__ __forceinline__ void wait_async() {
  asm volatile("s_wait_asynccnt 0x0" ::: "memory");
}
__device__ __forceinline__ unsigned lds_off(const void* p) {
  return (unsigned)(uintptr_t)p;   // generic LDS pointer: addr[31:0] = DS offset
}

// ---------------------------------------------------------------------------
// fp32 -> bf16 conversion (4 elements / thread)
// ---------------------------------------------------------------------------
__global__ __launch_bounds__(256) void cvt_f32_bf16(const float* __restrict__ in,
                                                    bf16* __restrict__ out) {
  int i = (blockIdx.x * 256 + threadIdx.x) * 4;
  float4 f = *(const float4*)(in + i);
  out[i + 0] = (bf16)f.x;
  out[i + 1] = (bf16)f.y;
  out[i + 2] = (bf16)f.z;
  out[i + 3] = (bf16)f.w;
}

// ---------------------------------------------------------------------------
// Tiled bf16 GEMM:  out[4096,2048] = X[4096,2048] * W[2048,2048]
// 128x128 tile / block (256 threads = 8 waves), each wave a 32x64 subtile.
// X tile staged via async global->LDS DMA; W tile transposed through VGPRs.
// STORE_F32: 0 -> bf16 output, 1 -> f32 output (compile-time).
// ---------------------------------------------------------------------------
template <int STORE_F32>
__global__ __launch_bounds__(256) void gemm_bf16(const bf16* __restrict__ X,
                                                 const bf16* __restrict__ W,
                                                 void* __restrict__ outp) {
  constexpr int GK = 2048;   // reduction dim
  constexpr int GN = 2048;   // output columns
  __shared__ bf16 Xs[128 * 40];   // row stride 80B (16B aligned)
  __shared__ bf16 Ws[128 * 40];   // transposed: Ws[n][k]

  const int tid  = threadIdx.x;
  const int wave = tid >> 5;
  const int lane = tid & 31;
  const int hf   = lane >> 4;   // lane half (0/1)
  const int l15  = lane & 15;

  const int tileN = blockIdx.x * 128;
  const int tileM = blockIdx.y * 128;
  const int waveM = (wave >> 1) * 32;
  const int waveN = (wave & 1) * 64;

  v8f acc[2][4] = {};

  for (int kb = 0; kb < GK / 32; ++kb) {
    __syncthreads();
    // --- stage X tile [128 x 32] via async DMA, 8B per lane per pass ---
#pragma unroll
    for (int p = 0; p < 4; ++p) {
      int idx = p * 256 + tid;
      int r = idx >> 3;
      int c = (idx & 7) * 4;
      async_ld_b64(lds_off(&Xs[r * 40 + c]),
                   &X[(size_t)(tileM + r) * GK + kb * 32 + c]);
    }
    // --- stage W tile transposed: Ws[n][k] = W[k][n] ---
#pragma unroll
    for (int p = 0; p < 4; ++p) {
      int idx = p * 256 + tid;
      int kr = idx >> 5;
      int c = (idx & 31) * 4;
      v4bf wv = *(const v4bf*)&W[(size_t)(kb * 32 + kr) * GN + tileN + c];
#pragma unroll
      for (int j = 0; j < 4; ++j) Ws[(c + j) * 40 + kr] = wv[j];
    }
    wait_async();
    __syncthreads();

    // --- fragments (ISA 16-bit A 16x32 / B 32x16 lane layouts) ---
    V16 afr[2], bfr[4];
#pragma unroll
    for (int i = 0; i < 2; ++i) {
      int row = waveM + i * 16 + l15;
      afr[i].h[0] = *(const v8bf*)&Xs[row * 40 + hf * 8];        // K 0-7 / 8-15
      afr[i].h[1] = *(const v8bf*)&Xs[row * 40 + hf * 8 + 16];   // K 16-23 / 24-31
    }
#pragma unroll
    for (int j = 0; j < 4; ++j) {
      int nl = waveN + j * 16 + l15;
      bfr[j].h[0] = *(const v8bf*)&Ws[nl * 40 + hf * 8];         // K 0-7 / 8-15
      bfr[j].h[1] = *(const v8bf*)&Ws[nl * 40 + hf * 8 + 16];    // K 16-23 / 24-31
    }
#pragma unroll
    for (int i = 0; i < 2; ++i)
#pragma unroll
      for (int j = 0; j < 4; ++j)
        acc[i][j] = wmma_bf16(afr[i].v, bfr[j].v, acc[i][j]);
  }

  // --- epilogue: C layout VGPR r -> M = r + 8*half, N = lane&15 ---
#pragma unroll
  for (int i = 0; i < 2; ++i)
#pragma unroll
    for (int j = 0; j < 4; ++j)
#pragma unroll
      for (int r = 0; r < 8; ++r) {
        int m = tileM + waveM + i * 16 + r + hf * 8;
        int n = tileN + waveN + j * 16 + l15;
        if constexpr (STORE_F32)
          ((float*)outp)[(size_t)m * GN + n] = acc[i][j][r];
        else
          ((bf16*)outp)[(size_t)m * GN + n] = (bf16)acc[i][j][r];
      }
}

// ---------------------------------------------------------------------------
// RoPE, in place on bf16 [B,S,N,H]; one thread per (b,s,n,h<64) pair.
// scale folds 1/sqrt(H) into Q.
// ---------------------------------------------------------------------------
__global__ __launch_bounds__(256) void rope_kernel(bf16* __restrict__ buf, float scale) {
  int idx = blockIdx.x * 256 + threadIdx.x;   // Bz*SEQ*NHEAD*64 threads
  int h = idx & 63;
  int n = (idx >> 6) & 15;
  int s = (idx >> 10) & 2047;
  int b = idx >> 21;
  bf16* p = buf + (((size_t)b * SEQ + s) * NHEAD + n) * HDIM;
  float ang = (float)s * __powf(1.0e4f, -(float)h * (1.0f / 64.0f));
  float sn, cs;
  __sincosf(ang, &sn, &cs);
  float q1 = (float)p[h];
  float q2 = (float)p[h + 64];
  p[h]      = (bf16)((q1 * cs - q2 * sn) * scale);
  p[h + 64] = (bf16)((q2 * cs + q1 * sn) * scale);
}

// ---------------------------------------------------------------------------
// Causal flash attention. Block = (qtile, head, batch), 256 threads = 8 waves.
// Each wave owns 16 query rows; loop over 32-key blocks up to causal limit.
// K tile staged via async global->LDS DMA; V transposed through VGPRs.
// ---------------------------------------------------------------------------
__global__ __launch_bounds__(256) void flash_kernel(const bf16* __restrict__ Q,
                                                    const bf16* __restrict__ Kb,
                                                    const bf16* __restrict__ Vb,
                                                    bf16* __restrict__ A) {
  __shared__ bf16 Ks[32 * 128];       // [key][h]
  __shared__ bf16 Vt[128 * 32];       // [h][key]  (transposed for B-fragments)
  __shared__ bf16 Ps[8 * 16 * 32];    // per-wave P staging [16 rows][32 keys]

  const int tid  = threadIdx.x;
  const int wave = tid >> 5;
  const int lane = tid & 31;
  const int hf   = lane >> 4;
  const int l15  = lane & 15;

  const int qt    = blockIdx.x;
  const int head  = blockIdx.y;
  const int b     = blockIdx.z;
  const int qrow0 = qt * 128 + wave * 16;

  // --- Q fragments for 16 rows x 128 depth (4 K-chunks of 32) ---
  V16 qf[4];
  const bf16* qp = Q + (((size_t)b * SEQ + (qrow0 + l15)) * NHEAD + head) * HDIM;
#pragma unroll
  for (int c = 0; c < 4; ++c) {
    qf[c].h[0] = *(const v8bf*)(qp + c * 32 + hf * 8);
    qf[c].h[1] = *(const v8bf*)(qp + c * 32 + hf * 8 + 16);
  }

  v8f o[8] = {};
  float mr[8], lr[8];
#pragma unroll
  for (int r = 0; r < 8; ++r) { mr[r] = -1e30f; lr[r] = 0.f; }

  const int nkb  = (qt + 1) * 4;      // causal: keys < (qt+1)*128
  const int srow = tid >> 3;          // 0..31
  const int scol = (tid & 7) * 16;    // 0..112
  bf16* pw = &Ps[wave * 512];

  for (int kb = 0; kb < nkb; ++kb) {
    const int tbase = kb * 32;
    __syncthreads();
    {  // stage K [32 keys x 128 h]: two async b128 DMAs per thread
      const bf16* src = Kb + (((size_t)b * SEQ + tbase + srow) * NHEAD + head) * HDIM + scol;
      async_ld_b128(lds_off(&Ks[srow * 128 + scol]), src);
      async_ld_b128(lds_off(&Ks[srow * 128 + scol + 8]), src + 8);
    }
    {  // stage V transposed: Vt[h][key]
      const bf16* src = Vb + (((size_t)b * SEQ + tbase + srow) * NHEAD + head) * HDIM + scol;
      v8bf a0 = *(const v8bf*)src;
      v8bf a1 = *(const v8bf*)(src + 8);
#pragma unroll
      for (int j = 0; j < 8; ++j) {
        Vt[(scol + j) * 32 + srow]     = a0[j];
        Vt[(scol + 8 + j) * 32 + srow] = a1[j];
      }
    }
    wait_async();
    __syncthreads();

    // --- S = Q * K^T for 16 rows x 32 keys (two 16-key C tiles) ---
    v8f s0 = {}, s1 = {};
#pragma unroll
    for (int c = 0; c < 4; ++c) {
      V16 b0, b1;
      b0.h[0] = *(const v8bf*)&Ks[l15 * 128 + c * 32 + hf * 8];
      b0.h[1] = *(const v8bf*)&Ks[l15 * 128 + c * 32 + hf * 8 + 16];
      b1.h[0] = *(const v8bf*)&Ks[(16 + l15) * 128 + c * 32 + hf * 8];
      b1.h[1] = *(const v8bf*)&Ks[(16 + l15) * 128 + c * 32 + hf * 8 + 16];
      s0 = wmma_bf16(qf[c].v, b0.v, s0);
      s1 = wmma_bf16(qf[c].v, b1.v, s1);
    }

    // --- online softmax (per row r, row m = r + 8*half, key = lane&15) ---
#pragma unroll
    for (int r = 0; r < 8; ++r) {
      const int m  = r + hf * 8;
      const int qg = qrow0 + m;
      float a0 = (tbase + l15      <= qg) ? s0[r] : -1e30f;
      float a1 = (tbase + 16 + l15 <= qg) ? s1[r] : -1e30f;
      float mx = fmaxf(a0, a1);
#pragma unroll
      for (int d = 1; d < 16; d <<= 1) mx = fmaxf(mx, __shfl_xor(mx, d, 32));
      const float mn = fmaxf(mr[r], mx);
      const float sc = __expf(mr[r] - mn);
      const float p0 = __expf(a0 - mn);
      const float p1 = __expf(a1 - mn);
      float rs = p0 + p1;
#pragma unroll
      for (int d = 1; d < 16; d <<= 1) rs += __shfl_xor(rs, d, 32);
      lr[r] = lr[r] * sc + rs;
      mr[r] = mn;
#pragma unroll
      for (int dd = 0; dd < 8; ++dd) o[dd][r] *= sc;   // rescale O rows
      pw[m * 32 + l15]      = (bf16)p0;                // stage P for A-fragment
      pw[m * 32 + 16 + l15] = (bf16)p1;
    }

    // --- O += P(16x32) * V(32x128) ---
    V16 pf;
    pf.h[0] = *(const v8bf*)&pw[l15 * 32 + hf * 8];
    pf.h[1] = *(const v8bf*)&pw[l15 * 32 + hf * 8 + 16];
#pragma unroll
    for (int dd = 0; dd < 8; ++dd) {
      V16 vf;
      vf.h[0] = *(const v8bf*)&Vt[(dd * 16 + l15) * 32 + hf * 8];
      vf.h[1] = *(const v8bf*)&Vt[(dd * 16 + l15) * 32 + hf * 8 + 16];
      o[dd] = wmma_bf16(pf.v, vf.v, o[dd]);
    }
  }

  // --- normalize and store A[b,s,n,h] (bf16, feeds output projection) ---
#pragma unroll
  for (int r = 0; r < 8; ++r) lr[r] = 1.f / lr[r];
#pragma unroll
  for (int dd = 0; dd < 8; ++dd)
#pragma unroll
    for (int r = 0; r < 8; ++r) {
      int m = r + hf * 8;
      int s = qrow0 + m;
      int h = dd * 16 + l15;
      A[(((size_t)b * SEQ + s) * NHEAD + head) * HDIM + h] = (bf16)(o[dd][r] * lr[r]);
    }
}

// ---------------------------------------------------------------------------
extern "C" void kernel_launch(void* const* d_in, const int* in_sizes, int n_in,
                              void* d_out, int out_size, void* d_ws, size_t ws_size,
                              hipStream_t stream) {
  const float* x_q  = (const float*)d_in[0];
  const float* x_kv = (const float*)d_in[1];
  const float* WQ   = (const float*)d_in[2];
  const float* WK   = (const float*)d_in[3];
  const float* WV   = (const float*)d_in[4];
  const float* WO   = (const float*)d_in[5];

  char* w = (char*)d_ws;
  size_t off = 0;
  auto carve = [&](size_t bytes) {
    void* p = w + off;
    off += (bytes + 255) & ~(size_t)255;
    return p;
  };
  bf16* xq_b  = (bf16*)carve((size_t)MROWS * EMB * 2);
  bf16* xkv_b = (bf16*)carve((size_t)MROWS * EMB * 2);
  bf16* wq_b  = (bf16*)carve((size_t)EMB * NHH * 2);
  bf16* wk_b  = (bf16*)carve((size_t)EMB * NHH * 2);
  bf16* wv_b  = (bf16*)carve((size_t)EMB * NHH * 2);
  bf16* wo_b  = (bf16*)carve((size_t)NHH * EMB * 2);
  bf16* Qb    = (bf16*)carve((size_t)MROWS * NHH * 2);
  bf16* Kb    = (bf16*)carve((size_t)MROWS * NHH * 2);
  bf16* Vb    = (bf16*)carve((size_t)MROWS * NHH * 2);
  bf16* Ab    = (bf16*)carve((size_t)MROWS * NHH * 2);

  // fp32 -> bf16
  cvt_f32_bf16<<<MROWS * EMB / 1024, 256, 0, stream>>>(x_q, xq_b);
  cvt_f32_bf16<<<MROWS * EMB / 1024, 256, 0, stream>>>(x_kv, xkv_b);
  cvt_f32_bf16<<<EMB * NHH / 1024, 256, 0, stream>>>(WQ, wq_b);
  cvt_f32_bf16<<<EMB * NHH / 1024, 256, 0, stream>>>(WK, wk_b);
  cvt_f32_bf16<<<EMB * NHH / 1024, 256, 0, stream>>>(WV, wv_b);
  cvt_f32_bf16<<<NHH * EMB / 1024, 256, 0, stream>>>(WO, wo_b);

  // Q/K/V projections (bf16 out)
  dim3 gproj(NHH / 128, MROWS / 128);
  gemm_bf16<0><<<gproj, 256, 0, stream>>>(xq_b,  wq_b, Qb);
  gemm_bf16<0><<<gproj, 256, 0, stream>>>(xkv_b, wk_b, Kb);
  gemm_bf16<0><<<gproj, 256, 0, stream>>>(xkv_b, wv_b, Vb);

  // RoPE (Q gets 1/sqrt(H) folded in)
  const int rope_threads = Bz * SEQ * NHEAD * 64;
  rope_kernel<<<rope_threads / 256, 256, 0, stream>>>(Qb, 0.08838834764831845f);
  rope_kernel<<<rope_threads / 256, 256, 0, stream>>>(Kb, 1.0f);

  // causal flash attention
  flash_kernel<<<dim3(SEQ / 128, NHEAD, Bz), 256, 0, stream>>>(Qb, Kb, Vb, Ab);

  // output projection -> fp32 d_out
  gemm_bf16<1><<<dim3(EMB / 128, MROWS / 128), 256, 0, stream>>>(Ab, wo_b, d_out);
}
